// MultiHeadCrossAttentionBlock_22402549416481
// MI455X (gfx1250) — matrix operand order repaired and checked by reading, not compiled, not run on hardware
//
#include <hip/hip_runtime.h>

// MI455X / gfx1250, wave32, WMMA f32_16x16x32_f16
// B=4, S=2048, D=512, H=8, DK=64

typedef _Float16 half_t;
typedef __attribute__((ext_vector_type(16))) _Float16 v16h;
typedef __attribute__((ext_vector_type(8)))  float    v8f;

union F16Frag {
    v16h   v;
    half_t h[16];
    float4 f4[2];
};

// A-matrix 16x32 f16 fragment from row-major [m][k] (ld = row stride in elems).
// Lane l: m = l%16, g = l/16; elems j<8 -> k = 8g+j ; j>=8 -> k = 16+8g+(j-8).
__device__ inline v16h load_a_frag(const half_t* base, int ld, int lane) {
    int m = lane & 15, g = lane >> 4;
    const half_t* p = base + (size_t)m * ld + 8 * g;
    F16Frag f;
    f.f4[0] = *(const float4*)(p);
    f.f4[1] = *(const float4*)(p + 16);
    return f.v;
}

// B-matrix 32x16 f16 fragment from row-major B^T [n][k] (ld = row stride).
// Lane l: n = l%16, g = l/16; elems j<8 -> k = 8g+j ; j>=8 -> k = 16+8g+(j-8).
__device__ inline v16h load_b_frag(const half_t* baseT, int ld, int lane) {
    int n = lane & 15, g = lane >> 4;
    const half_t* p = baseT + (size_t)n * ld + 8 * g;
    F16Frag f;
    f.f4[0] = *(const float4*)(p);
    f.f4[1] = *(const float4*)(p + 16);
    return f.v;
}

__device__ inline v8f wmma_f16(v16h a, v16h b, v8f c) {
    return __builtin_amdgcn_wmma_f32_16x16x32_f16(false, a, false, b, (short)0, c,
                                                  false, false);
}

// ---------------- conversion kernels ----------------

// WT layout: [WqT | WkT | WvT | WoutT], each 512x512 f16, WT_t[n][k] = W[k][n]
__global__ void conv_weights(const float* __restrict__ Wqkv,
                             const float* __restrict__ Wout,
                             half_t* __restrict__ WT) {
    int idx = blockIdx.x * 256 + threadIdx.x;
    if (idx >= 4 * 512 * 512) return;
    int t   = idx >> 18;              // which matrix
    int rem = idx & (512 * 512 - 1);
    int n = rem >> 9, k = rem & 511;
    float v = (t < 3) ? Wqkv[(size_t)k * 1536 + t * 512 + n]
                      : Wout[(size_t)k * 512 + n];
    WT[idx] = (half_t)v;
}

__global__ void conv_inputs(const float* __restrict__ Zq,
                            const float* __restrict__ Zkv,
                            half_t* __restrict__ Zq16,
                            half_t* __restrict__ Zkv16) {
    size_t idx = (size_t)blockIdx.x * 256 + threadIdx.x;
    const size_t N = (size_t)8192 * 512;
    if (idx < N)            Zq16[idx]      = (half_t)Zq[idx];
    else if (idx < 2 * N)   Zkv16[idx - N] = (half_t)Zkv[idx - N];
}

// ---------------- QKV projection ----------------
// grid: (B*S/16) * 96 tiles, 1 wave each. nt3: 0-31 -> Q, 32-63 -> K, 64-95 -> V
__global__ __launch_bounds__(32) void qkv_proj(const half_t* __restrict__ Zq16,
                                               const half_t* __restrict__ Zkv16,
                                               const half_t* __restrict__ WT,
                                               half_t* __restrict__ Q16,
                                               half_t* __restrict__ K16,
                                               half_t* __restrict__ Vt16) {
    int tile = blockIdx.x;
    int mt = tile / 96, nt3 = tile % 96;
    int third = nt3 >> 5, nc = nt3 & 31;
    int lane = threadIdx.x;
    const half_t* X    = (third == 0) ? Zq16 : Zkv16;
    const half_t* Wt   = WT + (size_t)third * 512 * 512 + (size_t)(nc * 16) * 512;
    const half_t* Arow = X + (size_t)mt * 16 * 512;
    v8f acc = {};
    for (int kc = 0; kc < 16; ++kc) {
        v16h a = load_a_frag(Arow + kc * 32, 512, lane);
        v16h b = load_b_frag(Wt   + kc * 32, 512, lane);
        acc = wmma_f16(a, b, acc);
    }
    int g = lane >> 4, n = lane & 15;
    int col = nc * 16 + n;            // 0..511 (tile never straddles a head)
    int h = col >> 6, dk = col & 63;
    for (int r = 0; r < 8; ++r) {
        int m  = mt * 16 + r + 8 * g;
        int b_ = m >> 11, s = m & 2047;
        half_t val = (half_t)acc[r];
        if (third == 0)
            Q16[((size_t)(b_ * 8 + h) * 2048 + s) * 64 + dk] = val;
        else if (third == 1)
            K16[((size_t)(b_ * 8 + h) * 2048 + s) * 64 + dk] = val;
        else
            Vt16[((size_t)(b_ * 8 + h) * 64 + dk) * 2048 + s] = val;
    }
}

// ---------------- attention ----------------
// grid: B*(S/16)=512 blocks, 128 threads (4 waves). Dynamic LDS (218 KB):
//   mean : 16 x 2048 f32   attn-mean accumulator over heads (written once)
//   scf  : 16 x 2056 f16   scores -> exp -> probs (padded row stride)
//   redM : 4 x 16 x 16 f32 per-wave per-lane row-max partials
//   red  : 16 x 8, rmax: 16, rinv: 16
//   ctxP : 4 x 16 x 64 f32 per-wave partial ctx (deterministic reduce)
#define SCLD16 2056
__global__ __launch_bounds__(128) void attention_f16(
        const half_t* __restrict__ Q16, const half_t* __restrict__ K16,
        const half_t* __restrict__ Vt16, const float* __restrict__ mask,
        half_t* __restrict__ ctx16, float* __restrict__ attn_mean) {
    extern __shared__ char smemraw[];
    float*  mean = (float*)smemraw;                                  // 131072 B
    half_t* scf  = (half_t*)(smemraw + 16 * 2048 * 4);               // 65792 B
    float*  redM = (float*)(smemraw + 16 * 2048 * 4 + 16 * SCLD16 * 2);
    float*  red  = redM + 4 * 16 * 16;
    float*  rmax = red + 16 * 8;
    float*  rinv = rmax + 16;
    float*  ctxP = rinv + 16;                                        // 4*16*64

    int blk = blockIdx.x;
    int b  = blk >> 7;
    int q0 = (blk & 127) * 16;
    int tid  = threadIdx.x;
    int wave = tid >> 5, lane = tid & 31;
    int g = lane >> 4, nn = lane & 15;
    int srow = tid >> 3, seg = tid & 7;

    for (int i = tid; i < 16 * 2048; i += 128) mean[i] = 0.f;

    for (int h = 0; h < 8; ++h) {
        __syncthreads();
        const half_t* Qb = Q16 + ((size_t)(b * 8 + h) * 2048 + q0) * 64;
        const half_t* Kb = K16 + (size_t)(b * 8 + h) * 2048 * 64;
        const half_t* Vb = Vt16 + (size_t)(b * 8 + h) * 64 * 2048;
        v16h aq0 = load_a_frag(Qb,      64, lane);
        v16h aq1 = load_a_frag(Qb + 32, 64, lane);

        // ---- scores: f16 into LDS, row max tracked in registers ----
        float rmx[8];
        for (int r = 0; r < 8; ++r) rmx[r] = -3.4e38f;
        for (int nt = wave; nt < 128; nt += 4) {
            int kv0 = nt * 16;
            if (nt + 4 < 128)
                __builtin_prefetch(Kb + (size_t)(kv0 + 64) * 64, 0, 0);
            v16h b0 = load_b_frag(Kb + (size_t)kv0 * 64,      64, lane);
            v16h b1 = load_b_frag(Kb + (size_t)kv0 * 64 + 32, 64, lane);
            v8f c = {};
            c = wmma_f16(aq0, b0, c);
            c = wmma_f16(aq1, b1, c);
            for (int r = 0; r < 8; ++r) {
                int row = r + 8 * g;
                float v = c[r] * 0.125f +
                          mask[(size_t)(q0 + row) * 2048 + kv0 + nn];
                half_t v16 = (half_t)v;
                rmx[r] = fmaxf(rmx[r], (float)v16);
                scf[row * SCLD16 + kv0 + nn] = v16;
            }
        }
        for (int r = 0; r < 8; ++r)
            redM[wave * 256 + (r + 8 * g) * 16 + nn] = rmx[r];
        __syncthreads();

        // ---- row max reduce (64 partials per row) ----
        {
            float m = -3.4e38f;
            for (int j = seg * 8; j < seg * 8 + 8; ++j)
                m = fmaxf(m, redM[(j >> 4) * 256 + srow * 16 + (j & 15)]);
            red[srow * 8 + seg] = m;
        }
        __syncthreads();
        if (tid < 16) {
            float m = red[tid * 8];
            for (int i = 1; i < 8; ++i) m = fmaxf(m, red[tid * 8 + i]);
            rmax[tid] = m;
        }
        __syncthreads();

        // ---- exp + partial row sum ----
        {
            float m = rmax[srow], s = 0.f;
            half_t* rp = scf + srow * SCLD16 + seg * 256;
            for (int i = 0; i < 256; ++i) {
                float e = __expf((float)rp[i] - m);
                rp[i] = (half_t)e;
                s += e;
            }
            red[srow * 8 + seg] = s;
        }
        __syncthreads();
        if (tid < 16) {
            float s = 0.f;
            for (int i = 0; i < 8; ++i) s += red[tid * 8 + i];
            rinv[tid] = 1.f / s;
        }
        __syncthreads();

        // ---- normalize probs (f16) + accumulate mean/H (f32) ----
        {
            float inv = rinv[srow];
            half_t* rp = scf + srow * SCLD16 + seg * 256;
            float*  mp = mean + srow * 2048 + seg * 256;
            for (int i = 0; i < 256; ++i) {
                float p = (float)rp[i] * inv;
                rp[i] = (half_t)p;
                mp[i] += p * 0.125f;   // 1/H
            }
        }
        __syncthreads();

        // ---- ctx = P @ V : wave w covers kv [w*512, w*512+512) ----
        // A-frag = two ds_load_b128 from f16 probs; B-frag = two global b128.
        for (int dkt = 0; dkt < 4; ++dkt) {
            v8f acc = {};
            const half_t* Vrow = Vb + (size_t)(dkt * 16 + nn) * 2048;
            const half_t* Pr   = scf + nn * SCLD16;
            for (int c = 0; c < 16; ++c) {
                int kvb = wave * 512 + c * 32;
                F16Frag af, bf;
                af.f4[0] = *(const float4*)(Pr + kvb + 8 * g);
                af.f4[1] = *(const float4*)(Pr + kvb + 16 + 8 * g);
                const half_t* pv = Vrow + kvb + 8 * g;
                bf.f4[0] = *(const float4*)(pv);
                bf.f4[1] = *(const float4*)(pv + 16);
                acc = wmma_f16(af.v, bf.v, acc);
            }
            for (int r = 0; r < 8; ++r)
                ctxP[wave * 1024 + (r + 8 * g) * 64 + dkt * 16 + nn] = acc[r];
        }
        __syncthreads();

        // ---- reduce 4 partials, write ctx16 [b*S+s][h*64+dk] ----
        for (int i = tid; i < 16 * 64; i += 128) {
            float v = ctxP[i] + ctxP[1024 + i] + ctxP[2048 + i] + ctxP[3072 + i];
            int row = i >> 6, dk = i & 63;
            ctx16[((size_t)(b * 2048 + q0 + row)) * 512 + h * 64 + dk] = (half_t)v;
        }
    }
    __syncthreads();
    for (int i = tid; i < 16 * 2048; i += 128) {
        int row = i >> 11, col = i & 2047;
        attn_mean[((size_t)(b * 2048 + q0 + row)) * 2048 + col] = mean[i];
    }
}

// ---------------- output projection ----------------
__global__ __launch_bounds__(32) void out_proj(const half_t* __restrict__ ctx16,
                                               const half_t* __restrict__ WoT,
                                               float* __restrict__ out) {
    int tile = blockIdx.x;            // (B*S/16) * 32
    int mt = tile >> 5, nt = tile & 31;
    int lane = threadIdx.x;
    const half_t* Arow = ctx16 + (size_t)mt * 16 * 512;
    const half_t* Bt   = WoT + (size_t)nt * 16 * 512;
    v8f acc = {};
    for (int kc = 0; kc < 16; ++kc) {
        v16h a = load_a_frag(Arow + kc * 32, 512, lane);
        v16h b = load_b_frag(Bt   + kc * 32, 512, lane);
        acc = wmma_f16(a, b, acc);
    }
    int g = lane >> 4, n = lane & 15;
    for (int r = 0; r < 8; ++r) {
        int m = mt * 16 + r + 8 * g;
        out[(size_t)m * 512 + nt * 16 + n] = acc[r];
    }
}

extern "C" void kernel_launch(void* const* d_in, const int* in_sizes, int n_in,
                              void* d_out, int out_size, void* d_ws, size_t ws_size,
                              hipStream_t stream) {
    (void)in_sizes; (void)n_in; (void)out_size; (void)ws_size;
    const float* Zq   = (const float*)d_in[0];
    const float* Zkv  = (const float*)d_in[1];
    const float* mask = (const float*)d_in[2];
    const float* Wqkv = (const float*)d_in[3];
    const float* Wout = (const float*)d_in[4];

    char* ws = (char*)d_ws;
    half_t* WT    = (half_t*)ws;                                   // 4*512*512
    half_t* Zq16  = (half_t*)(ws + (size_t)4 * 512 * 512 * 2);
    half_t* Zkv16 = Zq16  + (size_t)8192 * 512;
    half_t* Q16   = Zkv16 + (size_t)8192 * 512;
    half_t* K16   = Q16   + (size_t)4 * 8 * 2048 * 64;
    half_t* Vt16  = K16   + (size_t)4 * 8 * 2048 * 64;
    half_t* ctx16 = Vt16  + (size_t)4 * 8 * 64 * 2048;

    float* out       = (float*)d_out;
    float* attn_mean = out + (size_t)4 * 2048 * 512;

    conv_weights<<<(4 * 512 * 512 + 255) / 256, 256, 0, stream>>>(Wqkv, Wout, WT);
    conv_inputs<<<(int)((2ull * 8192 * 512 + 255) / 256), 256, 0, stream>>>(
        Zq, Zkv, Zq16, Zkv16);
    qkv_proj<<<512 * 96, 32, 0, stream>>>(Zq16, Zkv16, WT, Q16, K16, Vt16);

    size_t shmem = (size_t)16 * 2048 * 4 + (size_t)16 * SCLD16 * 2 +
                   (size_t)(4 * 16 * 16 + 16 * 8 + 16 + 16 + 4 * 16 * 64) * 4;
    (void)hipFuncSetAttribute(reinterpret_cast<const void*>(attention_f16),
                              hipFuncAttributeMaxDynamicSharedMemorySize,
                              (int)shmem);
    attention_f16<<<512, 128, shmem, stream>>>(Q16, K16, Vt16, mask, ctx16,
                                               attn_mean);

    out_proj<<<512 * 32, 32, 0, stream>>>(ctx16, WT + (size_t)3 * 512 * 512, out);
}
